// VladVQDirect_11879879544400
// MI455X (gfx1250) — matrix operand order repaired and checked by reading, not compiled, not run on hardware
//
#include <hip/hip_runtime.h>
#include <hip/hip_bf16.h>

// Problem constants (from reference)
#define TOK    8192   // B*S = 4*2048
#define DMODEL 1024
#define CBOOK  8192
#define NCENT  32

// GEMM tile config: 256 threads = 8 waves (wave32), waves 2x4, each wave 32x32 out
#define BM 64
#define BN 128
#define BK 32
#define APITCH 20   // dwords per packed A row (16 used + 4 pad; keeps 16B align, spreads banks)
#define BPITCH 20   // dwords per packed B row (n-major)

typedef __attribute__((ext_vector_type(16))) __bf16        v16bf;
typedef __attribute__((ext_vector_type(8)))  float         v8f;
typedef __attribute__((ext_vector_type(4)))  float         v4f;
typedef __attribute__((ext_vector_type(2)))  float         v2f;
typedef __attribute__((ext_vector_type(4)))  unsigned int  v4u;
typedef __attribute__((ext_vector_type(4)))  int           v4i;

#define GLOBAL_AS __attribute__((address_space(1)))
#define LDS_AS    __attribute__((address_space(3)))

#if __has_builtin(__builtin_amdgcn_global_load_async_to_lds_b128)
#define HAVE_ASYNC_LDS 1
#endif

__device__ __forceinline__ unsigned short f2bf_rne(float f) {
  unsigned int u = __float_as_uint(f);
  u += 0x7FFFu + ((u >> 16) & 1u);           // round-to-nearest-even to bf16
  return (unsigned short)(u >> 16);
}
__device__ __forceinline__ float bf2f(unsigned short s) {
  return __uint_as_float(((unsigned int)s) << 16);
}
// Split f0,f1 into packed bf16 hi-pair and lo-pair dwords
__device__ __forceinline__ void pack2(float f0, float f1, unsigned int& hi, unsigned int& lo) {
  const unsigned short h0 = f2bf_rne(f0);
  const unsigned short h1 = f2bf_rne(f1);
  const unsigned short l0 = f2bf_rne(f0 - bf2f(h0));
  const unsigned short l1 = f2bf_rne(f1 - bf2f(h1));
  hi = (unsigned int)h0 | ((unsigned int)h1 << 16);
  lo = (unsigned int)l0 | ((unsigned int)l1 << 16);
}

__device__ __forceinline__ void wait_async0() {
#if __has_builtin(__builtin_amdgcn_s_wait_asynccnt)
  __builtin_amdgcn_s_wait_asynccnt(0);
#else
  asm volatile("s_wait_asynccnt 0" ::: "memory");
#endif
}

union BFrag { v16bf v; v4u q[2]; };

// -----------------------------------------------------------------------------
// Kernel 1: logits[t, c] = x[t,:] . W[:,c] + b[c]  (written into encodings slot)
// bf16x3 split GEMM on v_wmma_f32_16x16x32_bf16, fragment-major LDS packing,
// double-buffered GLOBAL_LOAD_ASYNC_TO_LDS prefetch of the next K-tile.
// -----------------------------------------------------------------------------
__global__ __launch_bounds__(256) void vq_logits_gemm(
    const float* __restrict__ x,      // [TOK, DMODEL]
    const float* __restrict__ w,      // [DMODEL, CBOOK]
    const float* __restrict__ bias,   // [CBOOK]
    float* __restrict__ logits)       // [TOK, CBOOK]
{
  // Packed bf16 hi/lo tiles, fragment-major (pairs of K packed per dword)
  __shared__ unsigned int ahiP[BM * APITCH], aloP[BM * APITCH];   // 5 KB + 5 KB
  __shared__ unsigned int bhiP[BN * BPITCH], bloP[BN * BPITCH];   // 10 KB + 10 KB
#ifdef HAVE_ASYNC_LDS
  __shared__ float Araw[2][BM * BK];                              // 2 x 8 KB
  __shared__ float Braw[2][BK * BN];                              // 2 x 16 KB
#endif

  const int tid  = threadIdx.x;
  const int lane = tid & 31;
  const int wave = tid >> 5;
  const int g    = lane >> 4;        // lane half (0/1)
  const int ln   = lane & 15;
  const int m0   = blockIdx.y * BM;
  const int n0   = blockIdx.x * BN;
  const int wm   = (wave >> 2) * 32; // wave M offset: 0 or 32
  const int wn   = (wave & 3) * 32;  // wave N offset: 0..96

  v8f acc[2][2];
  for (int mt = 0; mt < 2; ++mt)
    for (int nt = 0; nt < 2; ++nt)
      acc[mt][nt] = (v8f){0.f, 0.f, 0.f, 0.f, 0.f, 0.f, 0.f, 0.f};

#ifdef HAVE_ASYNC_LDS
  // Issue async copy of raw f32 tile (k0) into Araw/Braw[buf] (16B per lane-op)
  auto issue_tile = [&](int k0, int buf) {
    #pragma unroll
    for (int t = 0; t < 2; ++t) {              // A tile: 2048 f32 = 512 x b128
      const int e  = (tid + t * 256) * 4;
      const int mm = e >> 5, kk = e & (BK - 1);
      __builtin_amdgcn_global_load_async_to_lds_b128(
          (GLOBAL_AS v4i*)(x + (size_t)(m0 + mm) * DMODEL + k0 + kk),
          (LDS_AS v4i*)(&Araw[buf][e]), 0, 0);
    }
    #pragma unroll
    for (int t = 0; t < 4; ++t) {              // B tile: 4096 f32 = 1024 x b128
      const int e  = (tid + t * 256) * 4;
      const int kk = e >> 7, nn = e & (BN - 1);
      __builtin_amdgcn_global_load_async_to_lds_b128(
          (GLOBAL_AS v4i*)(w + (size_t)(k0 + kk) * CBOOK + n0 + nn),
          (LDS_AS v4i*)(&Braw[buf][e]), 0, 0);
    }
  };
  int buf = 0;
  issue_tile(0, 0);
#endif

  for (int k0 = 0; k0 < DMODEL; k0 += BK) {
#ifdef HAVE_ASYNC_LDS
    wait_async0();               // this wave's raw tile (k0) landed in LDS
    __syncthreads();             // ... and every other wave's portion too
    if (k0 + BK < DMODEL) issue_tile(k0 + BK, buf ^ 1);   // overlap next tile

    // Convert raw f32 -> packed bf16 hi/lo (fragment-major)
    for (int i = tid; i < (BM * BK) / 2; i += 256) {       // A: 1024 K-pairs
      const int mm = i >> 4, kp = i & 15;
      const float f0 = Araw[buf][mm * BK + 2 * kp];
      const float f1 = Araw[buf][mm * BK + 2 * kp + 1];
      pack2(f0, f1, ahiP[mm * APITCH + kp], aloP[mm * APITCH + kp]);
    }
    for (int i = tid; i < (BK * BN) / 2; i += 256) {       // B: 2048 K-pairs
      const int nn = i & (BN - 1), kp = i >> 7;
      const float f0 = Braw[buf][(2 * kp) * BN + nn];
      const float f1 = Braw[buf][(2 * kp + 1) * BN + nn];
      pack2(f0, f1, bhiP[nn * BPITCH + kp], bloP[nn * BPITCH + kp]);
    }
    buf ^= 1;
#else
    // Direct path: f32 global loads -> packed bf16 hi/lo LDS
    for (int i = tid; i < (BM * BK) / 2; i += 256) {       // A: float2 loads
      const int mm = i >> 4, kp = i & 15;
      const v2f f = *(const v2f*)(x + (size_t)(m0 + mm) * DMODEL + k0 + 2 * kp);
      pack2(f.x, f.y, ahiP[mm * APITCH + kp], aloP[mm * APITCH + kp]);
    }
    for (int i = tid; i < (BK * BN) / 2; i += 256) {       // B: 2 coalesced rows
      const int nn = i & (BN - 1), kp = i >> 7;
      const float f0 = w[(size_t)(k0 + 2 * kp) * CBOOK + n0 + nn];
      const float f1 = w[(size_t)(k0 + 2 * kp + 1) * CBOOK + n0 + nn];
      pack2(f0, f1, bhiP[nn * BPITCH + kp], bloP[nn * BPITCH + kp]);
    }
    if (k0 + BK < DMODEL) {      // keep next tiles warm (global_prefetch_b8)
      __builtin_prefetch(&x[(size_t)(m0 + (tid >> 2)) * DMODEL + k0 + BK], 0, 1);
      __builtin_prefetch(&w[(size_t)(k0 + BK + (tid >> 3)) * CBOOK + n0], 0, 1);
    }
#endif
    __syncthreads();

    // Fragments: each hi/lo fragment = two 16B ds_load_b128 (fragment-major LDS)
    BFrag Ah[2], Al[2], Bh[2], Bl[2];
    #pragma unroll
    for (int mt = 0; mt < 2; ++mt) {
      const int m = wm + mt * 16 + ln;
      Ah[mt].q[0] = *(const v4u*)&ahiP[m * APITCH + 4 * g];
      Ah[mt].q[1] = *(const v4u*)&ahiP[m * APITCH + 8 + 4 * g];
      Al[mt].q[0] = *(const v4u*)&aloP[m * APITCH + 4 * g];
      Al[mt].q[1] = *(const v4u*)&aloP[m * APITCH + 8 + 4 * g];
    }
    #pragma unroll
    for (int nt = 0; nt < 2; ++nt) {
      const int n = wn + nt * 16 + ln;
      Bh[nt].q[0] = *(const v4u*)&bhiP[n * BPITCH + 4 * g];
      Bh[nt].q[1] = *(const v4u*)&bhiP[n * BPITCH + 8 + 4 * g];
      Bl[nt].q[0] = *(const v4u*)&bloP[n * BPITCH + 4 * g];
      Bl[nt].q[1] = *(const v4u*)&bloP[n * BPITCH + 8 + 4 * g];
    }

    // bf16x3: hi*hi + hi*lo + lo*hi (f32 accumulate)
    #pragma unroll
    for (int mt = 0; mt < 2; ++mt) {
      #pragma unroll
      for (int nt = 0; nt < 2; ++nt) {
        acc[mt][nt] = __builtin_amdgcn_wmma_f32_16x16x32_bf16(
            false, Ah[mt].v, false, Bh[nt].v, (short)0, acc[mt][nt], false, false);
        acc[mt][nt] = __builtin_amdgcn_wmma_f32_16x16x32_bf16(
            false, Ah[mt].v, false, Bl[nt].v, (short)0, acc[mt][nt], false, false);
        acc[mt][nt] = __builtin_amdgcn_wmma_f32_16x16x32_bf16(
            false, Al[mt].v, false, Bh[nt].v, (short)0, acc[mt][nt], false, false);
      }
    }
    __syncthreads();   // packed tiles are rewritten next iteration
  }

  // Epilogue: C/D layout is VGPR r -> M = r + 8*g, N = ln
  for (int mt = 0; mt < 2; ++mt) {
    for (int nt = 0; nt < 2; ++nt) {
      const int n = n0 + wn + nt * 16 + ln;
      const float bn = bias[n];
      #pragma unroll
      for (int r = 0; r < 8; ++r) {
        const int m = m0 + wm + mt * 16 + 8 * g + r;
        logits[(size_t)m * CBOOK + n] = acc[mt][nt][r] + bn;
      }
    }
  }
}

// -----------------------------------------------------------------------------
// Kernel 2: per token row — top-32 of logits (wave32 shfl-xor argmax),
// 32-way softmax (global denominator cancels under renormalization),
// in-place scatter into encodings, weighted codebook gather (L2-resident).
// -----------------------------------------------------------------------------
__global__ __launch_bounds__(256) void vq_topk_decode(
    const float* __restrict__ codebook,  // [CBOOK, DMODEL]
    float* __restrict__ enc,             // [TOK, CBOOK] in: logits, out: encodings
    float* __restrict__ quant,           // [TOK, DMODEL]
    float* __restrict__ oidx,            // [TOK, NCENT] (indices as float)
    float* __restrict__ ow)              // [TOK, NCENT]
{
  __shared__ float row[CBOOK];           // 32 KB
  __shared__ float wv[8];
  __shared__ int   wi[8];
  __shared__ int   s_idx[NCENT];
  __shared__ float s_val[NCENT];
  __shared__ float s_w[NCENT];

  const int t    = blockIdx.x;
  const int tid  = threadIdx.x;
  const int lane = tid & 31;
  const int wave = tid >> 5;
  const float NEG_INF = -__builtin_inff();

  {  // b128 row load
    const v4f* src = (const v4f*)(enc + (size_t)t * CBOOK);
    v4f* dst = (v4f*)row;
    for (int i = tid; i < CBOOK / 4; i += 256) dst[i] = src[i];
  }
  __syncthreads();

  // 32 rounds of argmax (ties -> lowest index, matching lax.top_k)
  for (int k = 0; k < NCENT; ++k) {
    float best = NEG_INF;
    int   bi   = CBOOK;
    for (int i = tid; i < CBOOK; i += 256) {
      const float v = row[i];
      if (v > best) { best = v; bi = i; }   // ascending i => lowest index wins ties
    }
    // wave32 shfl-xor reduction
    #pragma unroll
    for (int off = 16; off > 0; off >>= 1) {
      const float ov = __shfl_xor(best, off, 32);
      const int   oi = __shfl_xor(bi, off, 32);
      if (ov > best || (ov == best && oi < bi)) { best = ov; bi = oi; }
    }
    if (lane == 0) { wv[wave] = best; wi[wave] = bi; }
    __syncthreads();
    if (tid == 0) {
      float b = wv[0]; int j = wi[0];
      #pragma unroll
      for (int u = 1; u < 8; ++u) {
        if (wv[u] > b || (wv[u] == b && wi[u] < j)) { b = wv[u]; j = wi[u]; }
      }
      s_idx[k] = j;
      s_val[k] = b;
      row[j] = NEG_INF;                     // knock out winner
    }
    __syncthreads();
  }

  // Softmax over the 32 selected logits (== renormalized top softmax weights)
  if (tid == 0) {
    const float mx = s_val[0];
    float sum = 0.f;
    for (int k = 0; k < NCENT; ++k) {
      const float e = __expf(s_val[k] - mx);
      s_w[k] = e;
      sum += e;
    }
    const float inv = 1.f / sum;
    for (int k = 0; k < NCENT; ++k) s_w[k] *= inv;
  }
  __syncthreads();

  if (tid < NCENT) {
    oidx[(size_t)t * NCENT + tid] = (float)s_idx[tid];
    ow[(size_t)t * NCENT + tid]   = s_w[tid];
  }

  // Overwrite logits row with sparse encodings (top-k indices are distinct)
  {
    v4f* dst = (v4f*)(enc + (size_t)t * CBOOK);
    const v4f z = (v4f){0.f, 0.f, 0.f, 0.f};
    for (int i = tid; i < CBOOK / 4; i += 256) dst[i] = z;
  }
  __syncthreads();
  if (tid < NCENT)
    enc[(size_t)t * CBOOK + s_idx[tid]] = s_w[tid];

  // quantized[t, :] = sum_k w_k * codebook[idx_k, :]  (codebook fits in 192MB L2)
  {
    const v4f* cbv = (const v4f*)codebook;
    v4f a = (v4f){0.f, 0.f, 0.f, 0.f};
    #pragma unroll 8
    for (int k = 0; k < NCENT; ++k)
      a += cbv[(size_t)s_idx[k] * (DMODEL / 4) + tid] * s_w[k];
    ((v4f*)(quant + (size_t)t * DMODEL))[tid] = a;
  }
}

// -----------------------------------------------------------------------------
extern "C" void kernel_launch(void* const* d_in, const int* in_sizes, int n_in,
                              void* d_out, int out_size, void* d_ws, size_t ws_size,
                              hipStream_t stream) {
  const float* x    = (const float*)d_in[0];  // [4,2048,1024]
  const float* w    = (const float*)d_in[1];  // [1024,8192]
  const float* bias = (const float*)d_in[2];  // [8192]
  const float* cb   = (const float*)d_in[3];  // [8192,1024]

  float* out   = (float*)d_out;
  float* quant = out;                                   // TOK*DMODEL = 8388608
  float* oidx  = quant + (size_t)TOK * DMODEL;          // TOK*NCENT  = 262144
  float* ow    = oidx + (size_t)TOK * NCENT;            // TOK*NCENT  = 262144
  float* enc   = ow   + (size_t)TOK * NCENT;            // TOK*CBOOK  = 67108864

  dim3 gemm_grid(CBOOK / BN, TOK / BM);                 // (64, 128)
  vq_logits_gemm<<<gemm_grid, 256, 0, stream>>>(x, w, bias, enc);
  vq_topk_decode<<<TOK, 256, 0, stream>>>(cb, enc, quant, oidx, ow);
}
